// MultiHeadAttention_74543452389565
// MI455X (gfx1250) — matrix operand order repaired and checked by reading, not compiled, not run on hardware
//
#include <hip/hip_runtime.h>

// ---------------------------------------------------------------------------
// MI455X (gfx1250) fused causal multi-head attention forward.
//   B=2, S=2048, D=1024, H=16, hd=64.
// ~103 GFLOP vs ~100MB traffic -> compute bound by >100x at 23.3 TB/s, so all
// matmuls run on v_wmma_f32_16x16x32_bf16 (bf16 operands, fp32 accumulate).
// GEMM: 64x64 tile / 4 waves, 2x2 WMMA tiles per wave (fragment reuse),
// double-buffered GLOBAL_LOAD_ASYNC_TO_LDS_B128 staging (ASYNCcnt pipeline).
// Attention computes S^T = K*Q^T so softmax row-reduction over k is a single
// shfl_xor(16); V is stored transposed so PV B-fragments are contiguous
// ds_load_b128s. Workspace use: 48 MB.
// ---------------------------------------------------------------------------

#define SLEN   2048
#define DMODEL 1024
#define NH     16
#define HD     64

typedef __attribute__((ext_vector_type(16))) __bf16 bf16x16;
typedef __attribute__((ext_vector_type(8)))  float  f32x8;

union FragBF {
  uint4        q[2];   // two 16B chunks -> ds_load_b128
  unsigned int u[8];
  bf16x16      v;
};

static __device__ __forceinline__ unsigned short f2bf(float f) {
  unsigned int u = __float_as_uint(f);
  return (unsigned short)((u + 0x7FFFu + ((u >> 16) & 1u)) >> 16);  // RNE
}
static __device__ __forceinline__ float bf2f(unsigned short h) {
  return __uint_as_float(((unsigned int)h) << 16);
}

static __device__ __forceinline__ f32x8 wmma_bf16(bf16x16 a, bf16x16 b, f32x8 c) {
  // (neg_a, A, neg_b, B, c_mod, C, reuse_a, reuse_b)
  return __builtin_amdgcn_wmma_f32_16x16x32_bf16(false, a, false, b, (short)0, c,
                                                 false, false);
}

// Direct global->LDS async copy (16B per lane), tracked by ASYNCcnt.
// lds_off is the LDS byte offset = low 32 bits of the flat shared pointer
// (LDS aperture keeps the offset in addr[31:0]).
static __device__ __forceinline__ void async_copy_b128(unsigned lds_off,
                                                       unsigned long long gaddr) {
  asm volatile("global_load_async_to_lds_b128 %0, %1, off"
               :: "v"(lds_off), "v"(gaddr)
               : "memory");
}

template <int N>
static __device__ __forceinline__ void wait_asynccnt() {
#if __has_builtin(__builtin_amdgcn_s_wait_asynccnt)
  __builtin_amdgcn_s_wait_asynccnt(N);
#else
  asm volatile("s_wait_asynccnt %0" :: "i"(N) : "memory");
#endif
}

// ---------------------------------------------------------------------------
// fp32 -> bf16 conversion
// ---------------------------------------------------------------------------
__global__ void f32_to_bf16_kernel(const float* __restrict__ src,
                                   unsigned short* __restrict__ dst, int n) {
  int i = blockIdx.x * blockDim.x + threadIdx.x;
  if (i < n) dst[i] = f2bf(src[i]);
}

// ---------------------------------------------------------------------------
// bf16 WMMA GEMM:  C[M,N] = A[M,K] * Bw[N,K]^T   (i.e. X @ W^T)
// 64x64 output tile per 128-thread workgroup; each wave owns a 32x32 block
// computed as 2x2 WMMA tiles (A/B fragment reuse -> 4 WMMA per 8 ds_load_b128).
// K staged in 32-wide slabs via double-buffered async global->LDS copies.
// mode 0: store bf16 scattered to [b][h][s][hd]   (Q / K)
// mode 1: store bf16 scattered to [b][h][hd][s]   (V transposed)
// mode 2: store fp32 row-major [M][N]             (final projection)
// ---------------------------------------------------------------------------
__global__ void gemm_bf16_wmma(const unsigned short* __restrict__ A,
                               const unsigned short* __restrict__ Bw,
                               void* __restrict__ Cout,
                               int N, int Kdim, int mode)
{
  __shared__ uint4 At[2][64][4];   // [buf][row][16B chunk] : 64 x 32 bf16
  __shared__ uint4 Bt[2][64][4];   // W rows = B^T, N x K

  const int tid  = threadIdx.x;
  const int wave = tid >> 5;
  const int lane = tid & 31;
  const int hf   = lane >> 4;          // lane half (K-split of fragments)
  const int ln   = lane & 15;
  const int mi   = (wave & 1) * 32;
  const int ni   = (wave >> 1) * 32;

  const int colBase = blockIdx.x * 64;
  const int rowBase = blockIdx.y * 64;

  const uint4* Ag = (const uint4*)A;
  const uint4* Bg = (const uint4*)Bw;
  const int K8 = Kdim >> 3;            // uint4 per row

  f32x8 acc[2][2];
#pragma unroll
  for (int mt = 0; mt < 2; ++mt)
#pragma unroll
    for (int nt = 0; nt < 2; ++nt)
      acc[mt][nt] = f32x8{0.f, 0.f, 0.f, 0.f, 0.f, 0.f, 0.f, 0.f};

  // 4 async b128 copies per thread per stage (A: 2, B: 2) = 8KB per stage.
  auto issue = [&](int kb, int buf) {
    for (int i = tid; i < 256; i += 128) {
      int row = i >> 2, ch = i & 3;
      async_copy_b128((unsigned)(size_t)&At[buf][row][ch],
          (unsigned long long)(size_t)&Ag[(size_t)(rowBase + row) * K8 + kb * 4 + ch]);
      async_copy_b128((unsigned)(size_t)&Bt[buf][row][ch],
          (unsigned long long)(size_t)&Bg[(size_t)(colBase + row) * K8 + kb * 4 + ch]);
    }
  };

  const int nkb = Kdim >> 5;
  issue(0, 0);
  for (int kb = 0; kb < nkb; ++kb) {
    const int buf = kb & 1;
    if (kb + 1 < nkb) {          // prefetch next slab, allow its 4 loads pending
      issue(kb + 1, buf ^ 1);
      wait_asynccnt<4>();
    } else {
      wait_asynccnt<0>();
    }
    __syncthreads();
    FragBF a0, a1, b0, b1;
    // A fragment: row m, K elems {8*hf..8*hf+7, 16+8*hf..23+8*hf}
    a0.q[0] = At[buf][mi + ln][hf];          a0.q[1] = At[buf][mi + ln][2 + hf];
    a1.q[0] = At[buf][mi + 16 + ln][hf];     a1.q[1] = At[buf][mi + 16 + ln][2 + hf];
    // B fragment: col n, K elems {8*hf..8*hf+7, 16+8*hf..23+8*hf}
    b0.q[0] = Bt[buf][ni + ln][hf];      b0.q[1] = Bt[buf][ni + ln][2 + hf];
    b1.q[0] = Bt[buf][ni + 16 + ln][hf]; b1.q[1] = Bt[buf][ni + 16 + ln][2 + hf];
    acc[0][0] = wmma_bf16(a0.v, b0.v, acc[0][0]);
    acc[0][1] = wmma_bf16(a0.v, b1.v, acc[0][1]);
    acc[1][0] = wmma_bf16(a1.v, b0.v, acc[1][0]);
    acc[1][1] = wmma_bf16(a1.v, b1.v, acc[1][1]);
    __syncthreads();             // buffer being refilled was last read here
  }

#pragma unroll
  for (int mt = 0; mt < 2; ++mt) {
#pragma unroll
    for (int nt = 0; nt < 2; ++nt) {
      const int n = colBase + ni + nt * 16 + ln;
      if (mode == 2) {
        float* Cf = (float*)Cout;
#pragma unroll
        for (int r = 0; r < 8; ++r) {
          int R = rowBase + mi + mt * 16 + r + 8 * hf;
          Cf[(size_t)R * N + n] = acc[mt][nt][r];
        }
      } else {
        unsigned short* Cb = (unsigned short*)Cout;
        const int hcol = n >> 6, dh = n & 63;
#pragma unroll
        for (int r = 0; r < 8; ++r) {
          int R  = rowBase + mi + mt * 16 + r + 8 * hf;
          int bb = R >> 11, ss = R & (SLEN - 1);
          int idx = (mode == 0)
                        ? ((bb * NH + hcol) * SLEN + ss) * HD + dh   // [b][h][s][hd]
                        : ((bb * NH + hcol) * HD + dh) * SLEN + ss;  // [b][h][hd][s]
          Cb[idx] = f2bf(acc[mt][nt][r]);
        }
      }
    }
  }
}

// ---------------------------------------------------------------------------
// RoPE on Q/K in [b*h][s][64] layout. One thread per (bh, s, pair).
// ---------------------------------------------------------------------------
__global__ void rope_kernel(unsigned short* __restrict__ qk) {
  int t  = blockIdx.x * blockDim.x + threadIdx.x;   // 0 .. 2M-1
  int p  = t & 31;
  int s  = (t >> 5) & (SLEN - 1);
  int bh = t >> 16;                                  // /(32*2048)
  if (bh >= 2 * NH) return;
  int base = (bh * SLEN + s) * HD + 2 * p;
  float x1 = bf2f(qk[base]);
  float x2 = bf2f(qk[base + 1]);
  float theta = __powf(10000.0f, -(float)(2 * p) / 64.0f);
  float ang = (float)s * theta;
  float c = __cosf(ang), sn = __sinf(ang);
  qk[base]     = f2bf(x1 * c - x2 * sn);
  qk[base + 1] = f2bf(x1 * sn + x2 * c);
}

// ---------------------------------------------------------------------------
// Flash attention (causal). 128 threads = 4 waves per (b,h, 64-row q tile);
// each wave owns 16 q rows. k processed in 32-row blocks, double-buffered
// async global->LDS staging. Scores computed transposed (S^T = K*Q^T) so each
// lane's column is one q row: softmax stats need a single shfl_xor(16).
// O accumulated transposed (O^T = V^T * P^T) in 4 WMMA accumulators.
// ---------------------------------------------------------------------------
__global__ void flash_attn_kernel(const unsigned short* __restrict__ Q,
                                  const unsigned short* __restrict__ K,
                                  const unsigned short* __restrict__ Vt,
                                  unsigned short* __restrict__ O)
{
  __shared__ uint4 Kt[2][32][8];    // [buf] 32 k-rows x 64 bf16
  __shared__ uint4 Vtl[2][64][4];   // [buf] 64 dh-rows x 32 bf16 (V^T tile)

  const int tid  = threadIdx.x;
  const int wave = tid >> 5;
  const int lane = tid & 31;
  const int hf   = lane >> 4;
  const int ln   = lane & 15;

  const int qTile = blockIdx.x & 31;   // S/64 tiles
  const int bh    = blockIdx.x >> 5;   // 0..31
  const int b     = bh >> 4;
  const int h     = bh & 15;

  const int qBase = qTile * 64 + wave * 16;
  const int qg    = qBase + ln;        // this lane's query row

  const uint4* Qg = (const uint4*)Q;
  const uint4* Kg = (const uint4*)K;
  const uint4* Vg = (const uint4*)Vt;

  // Q as B-operand (B^T = Q rows): two K(=d) chunks of 32, resident in regs.
  FragBF qf0, qf1;
  {
    const size_t base = (size_t)(bh * SLEN + qBase + ln) * 8;  // uint4/row = 8
    qf0.q[0] = Qg[base + hf];
    qf0.q[1] = Qg[base + 2 + hf];
    qf1.q[0] = Qg[base + 4 + hf];
    qf1.q[1] = Qg[base + 6 + hf];
  }

  f32x8 acc0 = {0.f,0.f,0.f,0.f,0.f,0.f,0.f,0.f};
  f32x8 acc1 = acc0, acc2 = acc0, acc3 = acc0;
  float mrun = -3.0e38f;
  float lrun = 0.0f;

  // 4 async b128 copies per thread per stage (K: 2, V^T: 2) = 8KB per stage.
  auto issue = [&](int kb, int buf) {
    const int k0 = kb * 32;
    for (int i = tid; i < 256; i += 128) {
      int row = i >> 3, ch = i & 7;
      async_copy_b128((unsigned)(size_t)&Kt[buf][row][ch],
          (unsigned long long)(size_t)&Kg[(size_t)(bh * SLEN + k0 + row) * 8 + ch]);
    }
    for (int i = tid; i < 256; i += 128) {
      int row = i >> 2, ch = i & 3;
      async_copy_b128((unsigned)(size_t)&Vtl[buf][row][ch],
          (unsigned long long)(size_t)&Vg[(size_t)(bh * HD + row) * (SLEN / 8) + (k0 >> 3) + ch]);
    }
  };

  const int nB = qTile * 2 + 2;        // causal block count (uniform per WG)
  issue(0, 0);
  for (int kb = 0; kb < nB; ++kb) {
    const int k0  = kb * 32;
    const int buf = kb & 1;
    if (kb + 1 < nB) {
      issue(kb + 1, buf ^ 1);
      wait_asynccnt<4>();
    } else {
      wait_asynccnt<0>();
    }
    __syncthreads();
    if (k0 <= qBase + 15) {            // skip fully-masked blocks for this wave
      // S^T tiles (rows = k, cols = q), K-dim = d = 64 (2 chunks).
      FragBF ka;
      f32x8 st0 = {0.f,0.f,0.f,0.f,0.f,0.f,0.f,0.f};
      f32x8 st1 = st0;
      ka.q[0] = Kt[buf][ln][hf];          ka.q[1] = Kt[buf][ln][2 + hf];
      st0 = wmma_bf16(ka.v, qf0.v, st0);
      ka.q[0] = Kt[buf][ln][4 + hf];      ka.q[1] = Kt[buf][ln][6 + hf];
      st0 = wmma_bf16(ka.v, qf1.v, st0);
      ka.q[0] = Kt[buf][16 + ln][hf];     ka.q[1] = Kt[buf][16 + ln][2 + hf];
      st1 = wmma_bf16(ka.v, qf0.v, st1);
      ka.q[0] = Kt[buf][16 + ln][4 + hf]; ka.q[1] = Kt[buf][16 + ln][6 + hf];
      st1 = wmma_bf16(ka.v, qf1.v, st1);

      // Scale + causal mask + running softmax (per lane: q = ln).
      float s0[8], s1[8];
      float mloc = -3.0e38f;
#pragma unroll
      for (int r = 0; r < 8; ++r) {
        int kk0 = k0 + r + 8 * hf;       // tile0 k index
        int kk1 = kk0 + 16;              // tile1 k index
        float v0 = st0[r] * 0.125f;      // 1/sqrt(64)
        float v1 = st1[r] * 0.125f;
        if (kk0 > qg) v0 = -3.0e38f;
        if (kk1 > qg) v1 = -3.0e38f;
        s0[r] = v0; s1[r] = v1;
        mloc = fmaxf(mloc, fmaxf(v0, v1));
      }
      mloc = fmaxf(mloc, __shfl_xor(mloc, 16));
      const float mnew  = fmaxf(mrun, mloc);
      const float alpha = __expf(mrun - mnew);

      float lsum = 0.0f;
      unsigned int pk0[4], pk1[4];
#pragma unroll
      for (int j = 0; j < 4; ++j) {
        float pa = (s0[2*j]   <= -1.0e37f) ? 0.0f : __expf(s0[2*j]   - mnew);
        float pb = (s0[2*j+1] <= -1.0e37f) ? 0.0f : __expf(s0[2*j+1] - mnew);
        float pc = (s1[2*j]   <= -1.0e37f) ? 0.0f : __expf(s1[2*j]   - mnew);
        float pd = (s1[2*j+1] <= -1.0e37f) ? 0.0f : __expf(s1[2*j+1] - mnew);
        lsum += pa + pb + pc + pd;
        pk0[j] = (unsigned int)f2bf(pa) | ((unsigned int)f2bf(pb) << 16);
        pk1[j] = (unsigned int)f2bf(pc) | ((unsigned int)f2bf(pd) << 16);
      }
      lsum += __shfl_xor(lsum, 16);

      // Reassemble P into B-fragment layout (col n=q, K=k_local sequential,
      // lane-half selects k 0..15 vs 16..31): swap halves via shfl_xor(16).
      FragBF pf;
#pragma unroll
      for (int j = 0; j < 4; ++j) {
        unsigned int o0 = __shfl_xor(pk0[j], 16);
        unsigned int o1 = __shfl_xor(pk1[j], 16);
        pf.u[j]     = hf ? o1      : pk0[j];
        pf.u[4 + j] = hf ? pk1[j]  : o0;
      }

      // Rescale running O^T and accumulate V^T * P^T over 4 dh sub-tiles.
#pragma unroll
      for (int r = 0; r < 8; ++r) {
        acc0[r] *= alpha; acc1[r] *= alpha; acc2[r] *= alpha; acc3[r] *= alpha;
      }
      FragBF va;
      va.q[0] = Vtl[buf][ln][2 * hf];      va.q[1] = Vtl[buf][ln][2 * hf + 1];
      acc0 = wmma_bf16(va.v, pf.v, acc0);
      va.q[0] = Vtl[buf][16 + ln][2 * hf]; va.q[1] = Vtl[buf][16 + ln][2 * hf + 1];
      acc1 = wmma_bf16(va.v, pf.v, acc1);
      va.q[0] = Vtl[buf][32 + ln][2 * hf]; va.q[1] = Vtl[buf][32 + ln][2 * hf + 1];
      acc2 = wmma_bf16(va.v, pf.v, acc2);
      va.q[0] = Vtl[buf][48 + ln][2 * hf]; va.q[1] = Vtl[buf][48 + ln][2 * hf + 1];
      acc3 = wmma_bf16(va.v, pf.v, acc3);

      mrun = mnew;
      lrun = lrun * alpha + lsum;
    }
    __syncthreads();                   // refill target was last read here
  }

  // Store O row (q = ln), bf16 row-major [b*S+q][D] for the final projection.
  const float inv = 1.0f / lrun;
  unsigned short* dst =
      O + (size_t)(b * SLEN + qTile * 64 + wave * 16 + ln) * DMODEL + h * HD;
#pragma unroll
  for (int r = 0; r < 8; ++r) {
    dst[r + 8 * hf]      = f2bf(acc0[r] * inv);
    dst[16 + r + 8 * hf] = f2bf(acc1[r] * inv);
    dst[32 + r + 8 * hf] = f2bf(acc2[r] * inv);
    dst[48 + r + 8 * hf] = f2bf(acc3[r] * inv);
  }
}

// ---------------------------------------------------------------------------
// Launch: convert -> QKV GEMMs -> RoPE -> flash attention -> output GEMM.
// Workspace layout (48 MB): xbf 8M | wq/wk/wv/wo 2M each | Q 8M | K 8M |
// Vt 8M | O 8M.
// ---------------------------------------------------------------------------
extern "C" void kernel_launch(void* const* d_in, const int* in_sizes, int n_in,
                              void* d_out, int out_size, void* d_ws, size_t ws_size,
                              hipStream_t stream) {
  (void)in_sizes; (void)n_in; (void)out_size; (void)ws_size;
  const float* x  = (const float*)d_in[0];
  const float* Wq = (const float*)d_in[1];
  const float* Wk = (const float*)d_in[2];
  const float* Wv = (const float*)d_in[3];
  const float* Wo = (const float*)d_in[4];
  float* out = (float*)d_out;

  const int NX = 2 * SLEN * DMODEL;       // 4194304
  const int NW = DMODEL * DMODEL;         // 1048576

  char* p = (char*)d_ws;
  unsigned short* xbf = (unsigned short*)p; p += (size_t)NX * 2;
  unsigned short* wqb = (unsigned short*)p; p += (size_t)NW * 2;
  unsigned short* wkb = (unsigned short*)p; p += (size_t)NW * 2;
  unsigned short* wvb = (unsigned short*)p; p += (size_t)NW * 2;
  unsigned short* wob = (unsigned short*)p; p += (size_t)NW * 2;
  unsigned short* Qb  = (unsigned short*)p; p += (size_t)NX * 2;
  unsigned short* Kb  = (unsigned short*)p; p += (size_t)NX * 2;
  unsigned short* Vtb = (unsigned short*)p; p += (size_t)NX * 2;
  unsigned short* Ob  = (unsigned short*)p; p += (size_t)NX * 2;

  f32_to_bf16_kernel<<<NX / 256, 256, 0, stream>>>(x,  xbf, NX);
  f32_to_bf16_kernel<<<NW / 256, 256, 0, stream>>>(Wq, wqb, NW);
  f32_to_bf16_kernel<<<NW / 256, 256, 0, stream>>>(Wk, wkb, NW);
  f32_to_bf16_kernel<<<NW / 256, 256, 0, stream>>>(Wv, wvb, NW);
  f32_to_bf16_kernel<<<NW / 256, 256, 0, stream>>>(Wo, wob, NW);

  dim3 gg(DMODEL / 64, (2 * SLEN) / 64);  // (16, 64)
  dim3 gb(128);
  gemm_bf16_wmma<<<gg, gb, 0, stream>>>(xbf, wqb, Qb,  DMODEL, DMODEL, 0);
  gemm_bf16_wmma<<<gg, gb, 0, stream>>>(xbf, wkb, Kb,  DMODEL, DMODEL, 0);
  gemm_bf16_wmma<<<gg, gb, 0, stream>>>(xbf, wvb, Vtb, DMODEL, DMODEL, 1);

  rope_kernel<<<(2 * NH * SLEN * 32) / 256, 256, 0, stream>>>(Qb);
  rope_kernel<<<(2 * NH * SLEN * 32) / 256, 256, 0, stream>>>(Kb);

  flash_attn_kernel<<<2 * NH * (SLEN / 64), 128, 0, stream>>>(Qb, Kb, Vtb, Ob);

  gemm_bf16_wmma<<<gg, gb, 0, stream>>>(Ob, wob, out, DMODEL, DMODEL, 2);
}
